// MultiHeadAttention_68324339745328
// MI455X (gfx1250) — matrix operand order repaired and checked by reading, not compiled, not run on hardware
//
#include <hip/hip_runtime.h>
#include <hip/hip_bf16.h>

#define DEV __device__ __forceinline__

typedef __attribute__((ext_vector_type(16))) __bf16 bf16x16;
typedef __attribute__((ext_vector_type(8)))  float  floatx8;
typedef __attribute__((ext_vector_type(4)))  int    i32x4;

union ABFrag { bf16x16 v; uint4 q[2]; };

// ---- CDNA5 feature probes (compile-safe on both toolchains) ----------------
#if defined(__has_builtin)
#if __has_builtin(__builtin_amdgcn_global_load_async_to_lds_b128) && \
    __has_builtin(__builtin_amdgcn_s_wait_asynccnt)
#define HAS_ASYNC 1
#endif
#if __has_builtin(__builtin_amdgcn_cvt_pk_bf16_f32)
#define HAS_CVT_PK 1
#endif
#endif

DEV floatx8 zero8() {
    floatx8 z;
#pragma unroll
    for (int i = 0; i < 8; ++i) z[i] = 0.0f;
    return z;
}

DEV floatx8 wmma_bf16(bf16x16 a, bf16x16 b, floatx8 c) {
    return __builtin_amdgcn_wmma_f32_16x16x32_bf16(
        false, a, false, b, (short)0, c, false, false);
}

DEV unsigned short f2bf_sw(float f) {
    union { float f; unsigned u; } a; a.f = f;
    unsigned u = a.u + 0x7FFFu + ((a.u >> 16) & 1u);
    return (unsigned short)(u >> 16);
}

#ifdef HAS_CVT_PK
DEV unsigned pack2(float a, float b) {
    auto v = __builtin_amdgcn_cvt_pk_bf16_f32(a, b);
    return __builtin_bit_cast(unsigned, v);
}
#else
DEV unsigned pack2(float a, float b) {
    return (unsigned)f2bf_sw(a) | ((unsigned)f2bf_sw(b) << 16);
}
#endif

DEV unsigned short f2bf_raw(float f) {
    return (unsigned short)(pack2(f, 0.0f) & 0xFFFFu);
}

// global(generic) -> LDS 16-byte copy, async when available
DEV void cp16(const unsigned short* g, unsigned short* l) {
#ifdef HAS_ASYNC
    __builtin_amdgcn_global_load_async_to_lds_b128(
        (__attribute__((address_space(1))) i32x4*)g,
        (__attribute__((address_space(3))) i32x4*)l, 0, 0);
#else
    *(uint4*)l = *(const uint4*)g;
#endif
}

// Problem constants
static constexpr int D_MODEL = 1024;
static constexpr int H       = 16;
static constexpr int D_K     = 64;
static constexpr int B_SZ    = 2;
static constexpr int S_LEN   = 2048;
static constexpr int M_TOT   = B_SZ * S_LEN;      // 4096
static constexpr int NKSTEP  = D_MODEL / 32;      // 32 K-stages

static constexpr int XL_STRIDE = 40;   // 32 k + pad
static constexpr int WT_STRIDE = 72;   // 64 n + pad
static constexpr int PL_STRIDE = 72;   // 64 keys + pad

// ---------------------------------------------------------------------------
// Kernel: elementwise f32 -> bf16 (16 elems / thread)
// ---------------------------------------------------------------------------
__global__ __launch_bounds__(256)
void cvt_bf16_kernel(const float* __restrict__ X, unsigned short* __restrict__ Y) {
    const size_t i = ((size_t)blockIdx.x * 256 + threadIdx.x) * 16;
    const float4* s = (const float4*)(X + i);
    float4 f0 = s[0], f1 = s[1], f2 = s[2], f3 = s[3];
    uint4 o0 = make_uint4(pack2(f0.x, f0.y), pack2(f0.z, f0.w),
                          pack2(f1.x, f1.y), pack2(f1.z, f1.w));
    uint4 o1 = make_uint4(pack2(f2.x, f2.y), pack2(f2.z, f2.w),
                          pack2(f3.x, f3.y), pack2(f3.z, f3.w));
    uint4* d = (uint4*)(Y + i);
    d[0] = o0;
    d[1] = o1;
}

// ---------------------------------------------------------------------------
// Kernel: W[n][k] f32 -> Wt[k][n] bf16 (64x64 tile via LDS)
// ---------------------------------------------------------------------------
__global__ __launch_bounds__(256)
void wtrans_kernel(const float* __restrict__ W, unsigned short* __restrict__ Wt) {
    const int n0 = blockIdx.x * 64, k0 = blockIdx.y * 64;
    const int tid = threadIdx.x;
    __shared__ __align__(16) unsigned short Lt[64 * WT_STRIDE];

    {
        const int n = tid >> 2, c = tid & 3;
        const float4* src = (const float4*)(W + (size_t)(n0 + n) * D_MODEL + k0 + 16 * c);
        float4 f0 = src[0], f1 = src[1], f2 = src[2], f3 = src[3];
        unsigned short* base = Lt + (16 * c) * WT_STRIDE + n;
        base[ 0 * WT_STRIDE] = f2bf_sw(f0.x);
        base[ 1 * WT_STRIDE] = f2bf_sw(f0.y);
        base[ 2 * WT_STRIDE] = f2bf_sw(f0.z);
        base[ 3 * WT_STRIDE] = f2bf_sw(f0.w);
        base[ 4 * WT_STRIDE] = f2bf_sw(f1.x);
        base[ 5 * WT_STRIDE] = f2bf_sw(f1.y);
        base[ 6 * WT_STRIDE] = f2bf_sw(f1.z);
        base[ 7 * WT_STRIDE] = f2bf_sw(f1.w);
        base[ 8 * WT_STRIDE] = f2bf_sw(f2.x);
        base[ 9 * WT_STRIDE] = f2bf_sw(f2.y);
        base[10 * WT_STRIDE] = f2bf_sw(f2.z);
        base[11 * WT_STRIDE] = f2bf_sw(f2.w);
        base[12 * WT_STRIDE] = f2bf_sw(f3.x);
        base[13 * WT_STRIDE] = f2bf_sw(f3.y);
        base[14 * WT_STRIDE] = f2bf_sw(f3.z);
        base[15 * WT_STRIDE] = f2bf_sw(f3.w);
    }
    __syncthreads();
    {
        const int k = tid >> 2, seg = tid & 3;
        const uint4* s = (const uint4*)(Lt + k * WT_STRIDE + 16 * seg);
        uint4 a = s[0], b = s[1];
        uint4* d = (uint4*)(Wt + (size_t)(k0 + k) * D_MODEL + n0 + 16 * seg);
        d[0] = a;
        d[1] = b;
    }
}

// ---------------------------------------------------------------------------
// Kernel: bf16 GEMM  Y = Xb @ Wt  (Wt already [k][n]); 128x64 tile, 8 waves.
// Double-buffered async LDS staging (ASYNCcnt pipelined) when available.
// mode 0: bf16 head layout [B,H,S,64]   (Q/V projection)
// mode 1: bf16 transposed  [B,H,64,S]   (K projection)
// mode 2: f32  plain       [M, D_MODEL] (final output)
// ---------------------------------------------------------------------------
__global__ __launch_bounds__(256)
void gemm_bf16_kernel(const unsigned short* __restrict__ Xb,
                      const unsigned short* __restrict__ Wt,
                      void* __restrict__ Yout, int mode) {
    const int m0  = blockIdx.x * 128;
    const int n0  = blockIdx.y * 64;
    const int tid = threadIdx.x;
    const int w   = tid >> 5;
    const int L   = tid & 31;
    const int g   = L >> 4;
    const int lm  = L & 15;

    __shared__ __align__(16) unsigned short Xl[2][128 * XL_STRIDE];
    __shared__ __align__(16) unsigned short Wl[2][32 * WT_STRIDE];

    // 3 cp16 per thread per stage (2 for X tile, 1 for W tile)
    auto stage = [&](int kb, int buf) {
        const int m = tid >> 1, half = tid & 1;
        const unsigned short* gx = Xb + (size_t)(m0 + m) * D_MODEL + kb + 16 * half;
        unsigned short* lx = &Xl[buf][m * XL_STRIDE + 16 * half];
        cp16(gx, lx);
        cp16(gx + 8, lx + 8);
        const int row = tid >> 3, seg = tid & 7;
        cp16(Wt + (size_t)(kb + row) * D_MODEL + n0 + 8 * seg,
             &Wl[buf][row * WT_STRIDE + 8 * seg]);
    };

    floatx8 acc[4];
#pragma unroll
    for (int t = 0; t < 4; ++t) acc[t] = zero8();

    stage(0, 0);
    for (int i = 0; i < NKSTEP; ++i) {
        const int buf = i & 1;
        if (i + 1 < NKSTEP) {
            stage((i + 1) * 32, buf ^ 1);
#ifdef HAS_ASYNC
            __builtin_amdgcn_s_wait_asynccnt(3);   // current buf complete
#endif
        } else {
#ifdef HAS_ASYNC
            __builtin_amdgcn_s_wait_asynccnt(0);
#endif
        }
        __syncthreads();

        ABFrag a;
        a.q[0] = *(const uint4*)(&Xl[buf][(16 * w + lm) * XL_STRIDE + 8 * g]);
        a.q[1] = *(const uint4*)(&Xl[buf][(16 * w + lm) * XL_STRIDE + 16 + 8 * g]);
#pragma unroll
        for (int t = 0; t < 4; ++t) {
            ABFrag wf;
            const unsigned short* wc = &Wl[buf][16 * t + lm];
            unsigned wu[8];
#pragma unroll
            for (int j = 0; j < 8; ++j) {
                const int k0 = (j < 4) ? (8 * g + 2 * j) : (16 + 8 * g + 2 * (j - 4));
                wu[j] = (unsigned)wc[k0 * WT_STRIDE] | ((unsigned)wc[(k0 + 1) * WT_STRIDE] << 16);
            }
            wf.q[0] = make_uint4(wu[0], wu[1], wu[2], wu[3]);
            wf.q[1] = make_uint4(wu[4], wu[5], wu[6], wu[7]);
            acc[t] = wmma_bf16(a.v, wf.v, acc[t]);
        }
        __syncthreads();                            // readers done before overwrite
    }

#pragma unroll
    for (int t = 0; t < 4; ++t) {
#pragma unroll
        for (int r = 0; r < 8; ++r) {
            const int m = m0 + 16 * w + r + 8 * g;
            const int n = n0 + 16 * t + lm;
            if (mode == 2) {
                ((float*)Yout)[(size_t)m * D_MODEL + n] = acc[t][r];
            } else {
                const int bb = m >> 11, s = m & (S_LEN - 1);
                const int hh = n >> 6,  d = n & 63;
                unsigned short val = f2bf_raw(acc[t][r]);
                unsigned short* Y = (unsigned short*)Yout;
                if (mode == 0)
                    Y[(((size_t)bb * H + hh) * S_LEN + s) * D_K + d] = val;
                else
                    Y[(((size_t)bb * H + hh) * D_K + d) * S_LEN + s] = val;
            }
        }
    }
}

// ---------------------------------------------------------------------------
// Kernel: flash attention per (b,h). 128 threads = 4 waves x 16 q rows.
// Qh [bh][s][d], KhT [bh][d][s], Vh [bh][s][d] all bf16; ctx bf16 [b,s,h*64+d].
// ---------------------------------------------------------------------------
__global__ __launch_bounds__(128)
void attn_kernel(const unsigned short* __restrict__ Qh,
                 const unsigned short* __restrict__ KhT,
                 const unsigned short* __restrict__ Vh,
                 const int* __restrict__ mask,
                 unsigned short* __restrict__ ctx) {
    const int qb  = blockIdx.x;
    const int bh  = blockIdx.y;
    const int b   = bh >> 4, h = bh & 15;
    const int tid = threadIdx.x;
    const int w   = tid >> 5;
    const int L   = tid & 31;
    const int g   = L >> 4;
    const int lm  = L & 15;
    const int q0  = qb * 64 + 16 * w;

    __shared__ __align__(16) unsigned short Pl[4][16 * PL_STRIDE];

    // Q fragments in registers for the whole sweep
    ABFrag qa[2];
    {
        const unsigned short* qp = Qh + ((size_t)bh * S_LEN + q0 + lm) * D_K;
#pragma unroll
        for (int i = 0; i < 2; ++i) {
            qa[i].q[0] = *(const uint4*)(qp + 32 * i + 8 * g);
            qa[i].q[1] = *(const uint4*)(qp + 32 * i + 16 + 8 * g);
        }
    }

    floatx8 Oacc[4];
#pragma unroll
    for (int t = 0; t < 4; ++t) Oacc[t] = zero8();
    float mi[8], li[8];
#pragma unroll
    for (int r = 0; r < 8; ++r) { mi[r] = -1e30f; li[r] = 0.0f; }

    for (int kb = 0; kb < S_LEN; kb += 64) {
        __syncthreads();   // keep waves together; order Pl reuse across iters

        // prefetch next K/V tile rows into cache
        if (kb + 64 < S_LEN) {
            const int pr = tid & 63;
            if (tid < 64)
                __builtin_prefetch(KhT + ((size_t)bh * D_K + pr) * S_LEN + kb + 64, 0, 1);
            else
                __builtin_prefetch(Vh + ((size_t)bh * S_LEN + kb + 64 + pr) * D_K, 0, 1);
        }

        // ---- scores S = Q K^T * 0.125 (+mask): B-frags direct from KhT ----
        floatx8 S[4];
#pragma unroll
        for (int t = 0; t < 4; ++t) {
            floatx8 s = zero8();
#pragma unroll
            for (int i = 0; i < 2; ++i) {
                ABFrag kf;
                const unsigned short* kc =
                    KhT + ((size_t)bh * D_K + 32 * i) * S_LEN + kb + 16 * t + lm;
                unsigned ku[8];
#pragma unroll
                for (int j = 0; j < 8; ++j) {
                    const int k0 = (j < 4) ? (8 * g + 2 * j) : (16 + 8 * g + 2 * (j - 4));
                    ku[j] = (unsigned)kc[(size_t)k0 * S_LEN] | ((unsigned)kc[(size_t)(k0 + 1) * S_LEN] << 16);
                }
                kf.q[0] = make_uint4(ku[0], ku[1], ku[2], ku[3]);
                kf.q[1] = make_uint4(ku[4], ku[5], ku[6], ku[7]);
                s = wmma_bf16(qa[i].v, kf.v, s);
            }
            const int kp = kb + 16 * t + lm;
            const int mv = mask[b * S_LEN + kp];
#pragma unroll
            for (int r = 0; r < 8; ++r) {
                float sv = s[r] * 0.125f;
                s[r] = (mv == 0) ? -1e9f : sv;
            }
            S[t] = s;
        }

        // ---- online softmax (row = r + 8g; 16-lane half-wave reductions) ----
        float corr[8];
#pragma unroll
        for (int r = 0; r < 8; ++r) {
            float mx = fmaxf(fmaxf(S[0][r], S[1][r]), fmaxf(S[2][r], S[3][r]));
            mx = fmaxf(mx, __shfl_xor(mx, 1));
            mx = fmaxf(mx, __shfl_xor(mx, 2));
            mx = fmaxf(mx, __shfl_xor(mx, 4));
            mx = fmaxf(mx, __shfl_xor(mx, 8));
            const float mnew = fmaxf(mi[r], mx);
            const float c    = __expf(mi[r] - mnew);
            float rs = 0.0f;
#pragma unroll
            for (int t = 0; t < 4; ++t) {
                float p = __expf(S[t][r] - mnew);
                S[t][r] = p;
                rs += p;
            }
            rs += __shfl_xor(rs, 1);
            rs += __shfl_xor(rs, 2);
            rs += __shfl_xor(rs, 4);
            rs += __shfl_xor(rs, 8);
            li[r] = li[r] * c + rs;
            mi[r] = mnew;
            corr[r] = c;
        }
#pragma unroll
        for (int t = 0; t < 4; ++t)
#pragma unroll
            for (int r = 0; r < 8; ++r) Oacc[t][r] *= corr[r];

        // ---- P (C/D layout) -> per-wave LDS -> A-fragment layout ----
        unsigned short* pw = &Pl[w][0];
#pragma unroll
        for (int t = 0; t < 4; ++t)
#pragma unroll
            for (int r = 0; r < 8; ++r)
                pw[(r + 8 * g) * PL_STRIDE + 16 * t + lm] = f2bf_raw(S[t][r]);
        __syncthreads();

        // ---- O += P @ V (V B-frags direct from global, row-contiguous) ----
#pragma unroll
        for (int i = 0; i < 2; ++i) {
            ABFrag pa;
            pa.q[0] = *(const uint4*)(pw + lm * PL_STRIDE + 32 * i + 8 * g);
            pa.q[1] = *(const uint4*)(pw + lm * PL_STRIDE + 32 * i + 16 + 8 * g);
#pragma unroll
            for (int t = 0; t < 4; ++t) {
                ABFrag vf;
                const unsigned short* vc =
                    Vh + ((size_t)bh * S_LEN + kb + 32 * i) * D_K + 16 * t + lm;
                unsigned vu[8];
#pragma unroll
                for (int j = 0; j < 8; ++j) {
                    const int k0 = (j < 4) ? (8 * g + 2 * j) : (16 + 8 * g + 2 * (j - 4));
                    vu[j] = (unsigned)vc[k0 * D_K] | ((unsigned)vc[(k0 + 1) * D_K] << 16);
                }
                vf.q[0] = make_uint4(vu[0], vu[1], vu[2], vu[3]);
                vf.q[1] = make_uint4(vu[4], vu[5], vu[6], vu[7]);
                Oacc[t] = wmma_bf16(pa.v, vf.v, Oacc[t]);
            }
        }
    }

    // ---- epilogue ----
#pragma unroll
    for (int r = 0; r < 8; ++r) {
        const float inv = 1.0f / li[r];
        const int s = q0 + r + 8 * g;
#pragma unroll
        for (int t = 0; t < 4; ++t) {
            const float v = Oacc[t][r] * inv;
            ctx[((size_t)b * S_LEN + s) * D_MODEL + h * D_K + 16 * t + lm] =
                f2bf_raw(v);
        }
    }
}

// ---------------------------------------------------------------------------
extern "C" void kernel_launch(void* const* d_in, const int* in_sizes, int n_in,
                              void* d_out, int out_size, void* d_ws, size_t ws_size,
                              hipStream_t stream) {
    (void)in_sizes; (void)n_in; (void)out_size; (void)ws_size;

    const float* q    = (const float*)d_in[0];
    const float* k    = (const float*)d_in[1];
    const float* v    = (const float*)d_in[2];
    const int*   mask = (const int*)d_in[3];
    const float* Wq   = (const float*)d_in[4];
    const float* Wk   = (const float*)d_in[5];
    const float* Wv   = (const float*)d_in[6];
    const float* Wo   = (const float*)d_in[7];

    const size_t XE = (size_t)M_TOT * D_MODEL;      // 4M elems
    const size_t WE = (size_t)D_MODEL * D_MODEL;    // 1M elems
    unsigned short* p   = (unsigned short*)d_ws;
    unsigned short* Xq  = p;          p += XE;
    unsigned short* Xk  = p;          p += XE;
    unsigned short* Xv  = p;          p += XE;
    unsigned short* Wqt = p;          p += WE;
    unsigned short* Wkt = p;          p += WE;
    unsigned short* Wvt = p;          p += WE;
    unsigned short* Wot = p;          p += WE;
    unsigned short* Qh  = p;          p += XE;
    unsigned short* KhT = p;          p += XE;
    unsigned short* Vh  = p;          p += XE;
    unsigned short* ctx = p;          p += XE;

    const int cvtBlocks = (int)(XE / (16 * 256));   // 1024
    cvt_bf16_kernel<<<cvtBlocks, 256, 0, stream>>>(q, Xq);
    cvt_bf16_kernel<<<cvtBlocks, 256, 0, stream>>>(k, Xk);
    cvt_bf16_kernel<<<cvtBlocks, 256, 0, stream>>>(v, Xv);

    dim3 gW(D_MODEL / 64, D_MODEL / 64);            // 16 x 16
    wtrans_kernel<<<gW, 256, 0, stream>>>(Wq, Wqt);
    wtrans_kernel<<<gW, 256, 0, stream>>>(Wk, Wkt);
    wtrans_kernel<<<gW, 256, 0, stream>>>(Wv, Wvt);
    wtrans_kernel<<<gW, 256, 0, stream>>>(Wo, Wot);

    dim3 gG(M_TOT / 128, D_MODEL / 64);             // 32 x 16
    gemm_bf16_kernel<<<gG, 256, 0, stream>>>(Xq, Wqt, Qh, 0);
    gemm_bf16_kernel<<<gG, 256, 0, stream>>>(Xk, Wkt, KhT, 1);
    gemm_bf16_kernel<<<gG, 256, 0, stream>>>(Xv, Wvt, Vh, 0);

    dim3 gA(S_LEN / 64, B_SZ * H);                  // 32 x 32
    attn_kernel<<<gA, 128, 0, stream>>>(Qh, KhT, Vh, mask, ctx);

    gemm_bf16_kernel<<<gG, 256, 0, stream>>>(ctx, Wot, (float*)d_out, 2);
}
